// ScaledDotProductAttention_25744033973119
// MI455X (gfx1250) — matrix operand order repaired and checked by reading, not compiled, not run on hardware
//
#include <hip/hip_runtime.h>
#include <cstdint>
#include <cstddef>
#include <math.h>

typedef __attribute__((ext_vector_type(16))) _Float16 v16h;
typedef __attribute__((ext_vector_type(8)))  _Float16 v8h;
typedef __attribute__((ext_vector_type(2)))  _Float16 v2h;
typedef __attribute__((ext_vector_type(8)))  float    v8f;

#define B_  4
#define H_  16
#define S_  2048
#define D_  64
#define QT  128            // query rows per block (8 waves x 16)
#define KT  32             // keys per iteration
#define NKB (S_ / KT)      // 64 iterations
#define KHS 72             // K LDS row stride (halves): 144B, 16B-aligned, bank-skewed
#define VHS 40             // V^T LDS row stride (halves): 80B
#define PS  36             // P scratch row stride (floats)

// ---- gfx1250 async global->LDS (ASYNCcnt-tracked) ----
__device__ __forceinline__ void async_g2l_b32(uint32_t lds_off, uint64_t gaddr) {
    asm volatile("global_load_async_to_lds_b32 %0, %1, off"
                 :: "v"(lds_off), "v"(gaddr) : "memory");
}
__device__ __forceinline__ void wait_async0() {
    asm volatile("s_wait_asynccnt 0" ::: "memory");
}

__global__ __launch_bounds__(256)
void fa_fwd_kernel(const float* __restrict__ Q, const float* __restrict__ K,
                   const float* __restrict__ V, const int* __restrict__ mask,
                   float* __restrict__ O)
{
    __shared__ _Float16 skh[2][32 * KHS];   // K[key][d]   (f16, double-buffered)
    __shared__ _Float16 svh[2][64 * VHS];   // V^T[d][key] (f16, double-buffered)
    __shared__ float    spf[8][16 * PS];    // per-wave raw-score / P transpose scratch
    __shared__ int      smk[2][32];         // mask chunk (async DMA)

    const int tid  = threadIdx.x;
    const int lane = tid & 31;
    const int wave = tid >> 5;
    const int lh   = lane & 15;
    const int hi   = lane >> 4;
    // K staging map: 8 floats of one key row
    const int srow = tid >> 3;      // key row 0..31
    const int sseg = tid & 7;       // 8-float segment
    // V staging map: 2 adjacent keys x 4 d-values (enables packed b32 stores)
    const int vkp  = tid & 15;      // key pair -> keys {2vkp, 2vkp+1}
    const int vds  = tid >> 4;      // d segment -> d = vds*4 .. +3

    const int qb = blockIdx.x;
    const int bh = blockIdx.y;
    const int b  = bh / H_;
    const int q0 = qb * QT;

    const float* Qb = Q + (size_t)bh * S_ * D_;
    const float* Kb = K + (size_t)bh * S_ * D_;
    const float* Vb = V + (size_t)bh * S_ * D_;
    const int*   Mb = mask + (size_t)b * S_;
    float*       Ob = O + (size_t)bh * S_ * D_;

    // ---- Q tile as two 16x32 f16 A-fragments, pre-scaled by (1/sqrt(64))*log2(e)
    const float qscale = 0.125f * 1.44269504f;
    const float* qp = Qb + (size_t)(q0 + wave * 16 + lh) * D_;
    v16h aq[2];
#pragma unroll
    for (int c = 0; c < 2; ++c) {
        const int d0 = c * 32 + hi * 8;
#pragma unroll
        for (int i = 0; i < 8; ++i) {
            aq[c][i]     = (_Float16)(qp[d0 + i] * qscale);
            aq[c][8 + i] = (_Float16)(qp[d0 + 16 + i] * qscale);
        }
    }

    v8f acc[4];
#pragma unroll
    for (int t = 0; t < 4; ++t) acc[t] = (v8f){0.f,0.f,0.f,0.f,0.f,0.f,0.f,0.f};
    float mrunA = -INFINITY, lrunA = 0.f;   // A-layout stats: row lh per lane

    auto stage_mask = [&](int kb, int p) {
        if (wave == 0) {
            async_g2l_b32((uint32_t)(uintptr_t)&smk[p][lane],
                          (uint64_t)(uintptr_t)(Mb + kb * KT + lane));
        }
    };
    auto load_kv = [&](int kb, float4 kr[2], float4& v0, float4& v1) {
        const float* kp = Kb + (size_t)(kb * KT + srow) * D_ + sseg * 8;
        kr[0] = *(const float4*)kp;  kr[1] = *(const float4*)(kp + 4);
        const float* vp0 = Vb + (size_t)(kb * KT + 2 * vkp)     * D_ + vds * 4;
        const float* vp1 = Vb + (size_t)(kb * KT + 2 * vkp + 1) * D_ + vds * 4;
        v0 = *(const float4*)vp0;    v1 = *(const float4*)vp1;
    };
    auto store_stage = [&](int p, const float4 kr[2], const float4& v0, const float4& v1) {
        v8h hk;
        hk[0] = (_Float16)kr[0].x; hk[1] = (_Float16)kr[0].y;
        hk[2] = (_Float16)kr[0].z; hk[3] = (_Float16)kr[0].w;
        hk[4] = (_Float16)kr[1].x; hk[5] = (_Float16)kr[1].y;
        hk[6] = (_Float16)kr[1].z; hk[7] = (_Float16)kr[1].w;
        *(v8h*)&skh[p][srow * KHS + sseg * 8] = hk;
        // V^T: packed pairs {V[k][d], V[k+1][d]} -> one b32 each
        const float a0[4] = { v0.x, v0.y, v0.z, v0.w };
        const float a1[4] = { v1.x, v1.y, v1.z, v1.w };
#pragma unroll
        for (int i = 0; i < 4; ++i) {
            v2h h2; h2[0] = (_Float16)a0[i]; h2[1] = (_Float16)a1[i];
            *(v2h*)&svh[p][(vds * 4 + i) * VHS + 2 * vkp] = h2;
        }
    };

    // ---- prologue: stage chunk 0 into buffer 0 ----
    float4 kr[2], vr0, vr1;
    stage_mask(0, 0);
    load_kv(0, kr, vr0, vr1);
    store_stage(0, kr, vr0, vr1);

#pragma unroll 2
    for (int kb = 0; kb < NKB; ++kb) {
        const int p   = kb & 1;
        const int pn  = p ^ 1;
        const int kbn = (kb + 1 < NKB) ? kb + 1 : kb;

        wait_async0();
        __syncthreads();                 // buffer p ready; buffer pn free

        stage_mask(kbn, pn);
        load_kv(kbn, kr, vr0, vr1);      // overlaps compute below
        {
            const int kb2 = (kb + 2 < NKB) ? kb + 2 : kb;
            __builtin_prefetch(Kb + (size_t)(kb2 * KT + srow) * D_ + sseg * 8, 0, 3);
            __builtin_prefetch(Vb + (size_t)(kb2 * KT + 2 * vkp) * D_ + vds * 4, 0, 3);
        }

        // ================= compute on buffer p =================
        const _Float16* skp = skh[p];
        const _Float16* svp = svh[p];

        // S = Q*K^T : two 16x16 f32 tiles (keys 0..15 / 16..31)
        v8f s0 = (v8f){0.f,0.f,0.f,0.f,0.f,0.f,0.f,0.f};
        v8f s1 = s0;
#pragma unroll
        for (int c = 0; c < 2; ++c) {
            const _Float16* k0 = skp + lh * KHS        + c * 32 + hi * 8;
            const _Float16* k1 = skp + (16 + lh) * KHS + c * 32 + hi * 8;
            const v8h a0 = *(const v8h*)k0, a1 = *(const v8h*)(k0 + 16);
            const v8h b0 = *(const v8h*)k1, b1 = *(const v8h*)(k1 + 16);
            v16h bk0, bk1;
#pragma unroll
            for (int i = 0; i < 8; ++i) {
                bk0[i] = a0[i]; bk0[8 + i] = a1[i];
                bk1[i] = b0[i]; bk1[8 + i] = b1[i];
            }
            s0 = __builtin_amdgcn_wmma_f32_16x16x32_f16(false, aq[c], false, bk0,
                                                        (short)0, s0, false, false);
            s1 = __builtin_amdgcn_wmma_f32_16x16x32_f16(false, aq[c], false, bk1,
                                                        (short)0, s1, false, false);
        }

        // mask in C-layout, store RAW scores to per-wave LDS scratch
        const float NEGB = -1.44269504e10f;     // -1e10 * log2(e)
        const bool mk0 = (smk[p][lh] != 0);
        const bool mk1 = (smk[p][16 + lh] != 0);
        float* pw = spf[wave];
#pragma unroll
        for (int r = 0; r < 8; ++r) {
            const int m = r + hi * 8;
            pw[m * PS + lh]      = mk0 ? s0[r] : NEGB;
            pw[m * PS + 16 + lh] = mk1 ? s1[r] : NEGB;
        }

        // read back in A-layout: lane holds 16 keys of row lh (same-wave DS RAW)
        float tr[16];
        {
            const float* pr = pw + lh * PS + hi * 8;
#pragma unroll
            for (int i = 0; i < 8; ++i) { tr[i] = pr[i]; tr[8 + i] = pr[16 + i]; }
        }

        // row max: 15 in-register maxes + one xor-16 shuffle
        float rmax = tr[0];
#pragma unroll
        for (int i = 1; i < 16; ++i) rmax = fmaxf(rmax, tr[i]);
        rmax = fmaxf(rmax, __shfl_xor(rmax, 16, 32));

        // lazy rescale: wave-uniform branch, EXEC stays all-ones on both sides
        if (__ballot(rmax > mrunA)) {
            const float mnew   = fmaxf(mrunA, rmax);
            const float alphaA = exp2f(mrunA - mnew);
            mrunA  = mnew;
            lrunA *= alphaA;
            float alC[8];
#pragma unroll
            for (int r = 0; r < 8; ++r) alC[r] = __shfl(alphaA, r + 8 * hi, 32);
#pragma unroll
            for (int t = 0; t < 4; ++t)
#pragma unroll
                for (int r = 0; r < 8; ++r) acc[t][r] *= alC[r];
        }

        // exponentiate in A-layout, row sum, update l
        float pv[16];
        float rsum = 0.f;
#pragma unroll
        for (int i = 0; i < 16; ++i) { pv[i] = exp2f(tr[i] - mrunA); rsum += pv[i]; }
        rsum += __shfl_xor(rsum, 16, 32);
        lrunA += rsum;

        // P fragment: already in A-layout, convert to f16
        v16h ap;
#pragma unroll
        for (int i = 0; i < 16; ++i) ap[i] = (_Float16)pv[i];

        // O += P * V : 4 WMMAs over 64-wide d
#pragma unroll
        for (int t = 0; t < 4; ++t) {
            const _Float16* vp = svp + (t * 16 + lh) * VHS + hi * 8;
            const v8h c0 = *(const v8h*)vp, c1 = *(const v8h*)(vp + 16);
            v16h bv;
#pragma unroll
            for (int i = 0; i < 8; ++i) { bv[i] = c0[i]; bv[8 + i] = c1[i]; }
            acc[t] = __builtin_amdgcn_wmma_f32_16x16x32_f16(false, ap, false, bv,
                                                            (short)0, acc[t], false, false);
        }
        // =======================================================

        store_stage(pn, kr, vr0, vr1);   // convert + store next chunk
    }

    // ---- epilogue: per-row l back to C-layout, normalize, store ----
    float invC[8];
#pragma unroll
    for (int r = 0; r < 8; ++r) invC[r] = 1.0f / __shfl(lrunA, r + 8 * hi, 32);
#pragma unroll
    for (int t = 0; t < 4; ++t)
#pragma unroll
        for (int r = 0; r < 8; ++r) {
            const int m = r + hi * 8;
            Ob[(size_t)(q0 + wave * 16 + m) * D_ + t * 16 + lh] = acc[t][r] * invC[r];
        }
}

extern "C" void kernel_launch(void* const* d_in, const int* in_sizes, int n_in,
                              void* d_out, int out_size, void* d_ws, size_t ws_size,
                              hipStream_t stream) {
    (void)in_sizes; (void)n_in; (void)out_size; (void)d_ws; (void)ws_size;
    const float* Q = (const float*)d_in[0];
    const float* K = (const float*)d_in[1];
    const float* V = (const float*)d_in[2];
    const int*   M = (const int*)d_in[3];
    float*       O = (float*)d_out;

    dim3 grid(S_ / QT, B_ * H_);
    fa_fwd_kernel<<<grid, dim3(256), 0, stream>>>(Q, K, V, M, O);
}
